// GAT_Yelp_13606456394534
// MI455X (gfx1250) — matrix operand order repaired and checked by reading, not compiled, not run on hardware
//
#include <hip/hip_runtime.h>

typedef __attribute__((ext_vector_type(16))) __bf16 v16bf;
typedef __attribute__((ext_vector_type(8)))  float  v8f;

#define NEG_SLOPE 0.2f

// ---------- helpers ----------
static __device__ __forceinline__ unsigned short f2bf(float f) {
  unsigned u = __float_as_uint(f);
  u += 0x7FFFu + ((u >> 16) & 1u);          // round to nearest even
  return (unsigned short)(u >> 16);
}
// monotone float<->uint map so unsigned atomicMax == float max
static __device__ __forceinline__ unsigned fflip(float f) {
  unsigned u = __float_as_uint(f);
  return (u >> 31) ? ~u : (u | 0x80000000u);
}
static __device__ __forceinline__ float funflip(unsigned u) {
  unsigned v = (u >> 31) ? (u & 0x7FFFFFFFu) : ~u;
  return __uint_as_float(v);
}

// ---------- elementwise kernels ----------
__global__ void fill_zero_kernel(unsigned* __restrict__ p, long n) {
  long i = (long)blockIdx.x * blockDim.x + threadIdx.x;
  if (i < n) p[i] = 0u;
}

// fp32 [N,Kin] row-major -> bf16 [N,Kpad] row-major (zero padded K)
__global__ void cvt_node_bf16_kernel(const float* __restrict__ in,
                                     unsigned short* __restrict__ out,
                                     int N, int Kin, int Kpad) {
  long i = (long)blockIdx.x * blockDim.x + threadIdx.x;
  long total = (long)N * Kpad;
  if (i >= total) return;
  int  k = (int)(i % Kpad);
  long n = i / Kpad;
  float v = (k < Kin) ? in[n * (long)Kin + k] : 0.0f;
  out[i] = f2bf(v);
}

// fp32 W [Kin,Nin] row-major -> bf16 Wt [Npad,Kpad] (column-major for WMMA B frag)
__global__ void cvt_w_bf16t_kernel(const float* __restrict__ W,
                                   unsigned short* __restrict__ out,
                                   int Kin, int Nin, int Kpad, int Npad) {
  long i = (long)blockIdx.x * blockDim.x + threadIdx.x;
  long total = (long)Npad * Kpad;
  if (i >= total) return;
  int k = (int)(i % Kpad);
  int n = (int)(i / Kpad);
  float v = (k < Kin && n < Nin) ? W[(long)k * Nin + n] : 0.0f;
  out[i] = f2bf(v);
}

// ---------- WMMA GEMM: C[M,Npad] = A[M,Kpad](bf16) * B^T[Npad,Kpad](bf16) ----------
__global__ void __launch_bounds__(256)
gemm_bf16_wmma_kernel(const unsigned short* __restrict__ A,
                      const unsigned short* __restrict__ B,
                      float* __restrict__ C, int M, int Npad, int Kpad) {
  const int lane  = threadIdx.x & 31;
  const int wave  = threadIdx.x >> 5;
  const int mBase = (blockIdx.x * 8 + wave) * 16;
  if (mBase >= M) return;                       // wave-uniform: EXEC stays all-1s
  const int nBase = blockIdx.y * 16;
  const int half  = lane >> 4;                  // 0: lanes 0-15, 1: lanes 16-31
  const int l16   = lane & 15;

  int arow = mBase + l16;                       // A row for this lane (clamped dup ok)
  if (arow >= M) arow = M - 1;
  const unsigned short* Ap = A + (long)arow * Kpad + half * 8;          // K runs
  const unsigned short* Bp = B + (long)(nBase + l16) * Kpad + half * 8;  // K runs

  v8f c = {0.f, 0.f, 0.f, 0.f, 0.f, 0.f, 0.f, 0.f};
  for (int k = 0; k < Kpad; k += 32) {
    union { v16bf v; uint4 q[2]; } a, b;
    a.q[0] = *(const uint4*)(Ap + k);           // K = k + half*8 + [0..7]
    a.q[1] = *(const uint4*)(Ap + k + 16);      // K = k + 16 + half*8 + [0..7]
    b.q[0] = *(const uint4*)(Bp + k);           // K = k + half*8 + [0..7]
    b.q[1] = *(const uint4*)(Bp + k + 16);      // K = k + 16 + half*8 + [0..7]
    c = __builtin_amdgcn_wmma_f32_16x16x32_bf16(false, a.v, false, b.v,
                                                (short)0, c, false, false);
  }
  const int ncol = nBase + l16;
#pragma unroll
  for (int r = 0; r < 8; ++r) {                 // VGPR r -> row (half*8 + r)
    int row = mBase + half * 8 + r;
    if (row < M) C[(long)row * Npad + ncol] = c[r];
  }
}

// ---------- per-node attention logits: SL[n,h] = <h[n,h,:], a_src[h,:]>, same for SD ----------
__global__ void __launch_bounds__(256)
node_logits_kernel(const float* __restrict__ Hlin,
                   const float* __restrict__ a_s, const float* __restrict__ a_d,
                   float* __restrict__ SL, float* __restrict__ SD,
                   int N, int Npad, int Hh, int C) {
  const int lane = threadIdx.x & 31;
  long wid = (long)blockIdx.x * (blockDim.x >> 5) + (threadIdx.x >> 5);
  long node = wid / Hh;
  int  h    = (int)(wid % Hh);
  if (node >= N) return;
  const float* hp = Hlin + node * (long)Npad + h * C;
  float ss = 0.f, sd = 0.f;
  for (int cix = lane; cix < C; cix += 32) {
    float v = hp[cix];
    ss += v * a_s[h * C + cix];
    sd += v * a_d[h * C + cix];
  }
  for (int o = 16; o > 0; o >>= 1) {
    ss += __shfl_down(ss, o, 32);
    sd += __shfl_down(sd, o, 32);
  }
  if (lane == 0) { SL[node * Hh + h] = ss; SD[node * Hh + h] = sd; }
}

// ---------- edge pass A: segment max (flipped-uint atomicMax) ----------
__global__ void edge_max_kernel(const int* __restrict__ ei, long E, long Etot,
                                const float* __restrict__ SL, const float* __restrict__ SD,
                                unsigned* __restrict__ Mx, int Hh) {
  long i = (long)blockIdx.x * blockDim.x + threadIdx.x;
  if (i >= Etot * Hh) return;
  long e = i / Hh; int h = (int)(i % Hh);
  int s, d;
  if (e < E) { s = ei[e]; d = ei[E + e]; } else { s = (int)(e - E); d = s; }
  float v = SL[(long)s * Hh + h] + SD[(long)d * Hh + h];
  v = (v > 0.f) ? v : NEG_SLOPE * v;            // leaky_relu
  atomicMax(&Mx[(long)d * Hh + h], fflip(v));
}

// ---------- edge pass B: ex = exp(e - max), denom += ex ----------
__global__ void edge_expsum_kernel(const int* __restrict__ ei, long E, long Etot,
                                   const float* __restrict__ SL, const float* __restrict__ SD,
                                   const unsigned* __restrict__ Mx,
                                   float* __restrict__ DEN, float* __restrict__ EX, int Hh) {
  long i = (long)blockIdx.x * blockDim.x + threadIdx.x;
  if (i >= Etot * Hh) return;
  long e = i / Hh; int h = (int)(i % Hh);
  int s, d;
  if (e < E) { s = ei[e]; d = ei[E + e]; } else { s = (int)(e - E); d = s; }
  float v = SL[(long)s * Hh + h] + SD[(long)d * Hh + h];
  v = (v > 0.f) ? v : NEG_SLOPE * v;
  float ex = __expf(v - funflip(Mx[(long)d * Hh + h]));
  EX[e * Hh + h] = ex;
  atomicAdd(&DEN[(long)d * Hh + h], ex);
}

// ---------- edge pass C: out[dst] += alpha * h[src]  (one wave per edge) ----------
__global__ void __launch_bounds__(256)
edge_agg_kernel(const int* __restrict__ ei, long E, long Etot,
                const float* __restrict__ Hlin, int Npad, int nvalid, int C, int Hh,
                const float* __restrict__ EX, const float* __restrict__ DEN,
                float* __restrict__ Out, int OutStride) {
  const int lane = threadIdx.x & 31;
  long e = (long)blockIdx.x * (blockDim.x >> 5) + (threadIdx.x >> 5);
  if (e >= Etot) return;
  int s, d;
  if (e < E) { s = ei[e]; d = ei[E + e]; } else { s = (int)(e - E); d = s; }
  const float* hs = Hlin + (long)s * Npad;
  float*       od = Out + (long)d * OutStride;
  for (int f = lane; f < nvalid; f += 32) {
    int h = f / C;
    float alpha = EX[e * Hh + h] / (DEN[(long)d * Hh + h] + 1e-16f);
    atomicAdd(&od[f], alpha * hs[f]);
  }
}

// ---------- bias + ELU, emit bf16 for the next layer's GEMM ----------
__global__ void bias_elu_cvt_kernel(const float* __restrict__ G, const float* __restrict__ b,
                                    unsigned short* __restrict__ A, int N, int F) {
  long i = (long)blockIdx.x * blockDim.x + threadIdx.x;
  if (i >= (long)N * F) return;
  float v = G[i] + b[i % F];
  v = (v > 0.f) ? v : (__expf(v) - 1.0f);       // elu
  A[i] = f2bf(v);
}

__global__ void bias_add_kernel(float* __restrict__ O, const float* __restrict__ b,
                                int N, int F) {
  long i = (long)blockIdx.x * blockDim.x + threadIdx.x;
  if (i >= (long)N * F) return;
  O[i] += b[i % F];
}

// ---------- host launcher ----------
extern "C" void kernel_launch(void* const* d_in, const int* in_sizes, int n_in,
                              void* d_out, int out_size, void* d_ws, size_t ws_size,
                              hipStream_t stream) {
  (void)n_in; (void)ws_size;
  const float* x   = (const float*)d_in[0];
  const int*   ei  = (const int*)  d_in[1];
  const float* W1  = (const float*)d_in[2];
  const float* a1s = (const float*)d_in[3];
  const float* a1d = (const float*)d_in[4];
  const float* b1  = (const float*)d_in[5];
  const float* W2  = (const float*)d_in[6];
  const float* a2s = (const float*)d_in[7];
  const float* a2d = (const float*)d_in[8];
  const float* b2  = (const float*)d_in[9];
  const float* W3  = (const float*)d_in[10];
  const float* a3s = (const float*)d_in[11];
  const float* a3d = (const float*)d_in[12];
  const float* b3  = (const float*)d_in[13];

  const int FIN = 300, FINP = 320, HID = 256, OUT = 100, OUTP = 112;
  const int  N    = in_sizes[0] / FIN;
  const long E    = (long)in_sizes[1] / 2;
  const long Etot = E + N;

  // workspace partition (~143 MB)
  char* w = (char*)d_ws; size_t off = 0;
  auto take = [&](size_t bytes) -> char* {
    char* p = w + off; off += (bytes + 255) & ~(size_t)255; return p;
  };
  unsigned short* Abuf = (unsigned short*)take((size_t)N * FINP * 2); // bf16 node feats
  float*          Hbuf = (float*)take((size_t)N * HID * 4);           // GEMM output h_lin
  float*          Gbuf = (float*)take((size_t)N * HID * 4);           // aggregated output
  unsigned short* Wb   = (unsigned short*)take((size_t)FINP * HID * 2);
  float*    SL  = (float*)take((size_t)N * 2 * 4);
  float*    SD  = (float*)take((size_t)N * 2 * 4);
  unsigned* Mx  = (unsigned*)take((size_t)N * 2 * 4);
  float*    DEN = (float*)take((size_t)N * 2 * 4);
  float*    EXb = (float*)take((size_t)Etot * 2 * 4);

  const int TB = 256;
  auto nb = [](long n, int per) { return (unsigned)((n + per - 1) / per); };

  auto edge_softmax_agg = [&](const float* Hlin, int Npad, int nvalid, int C, int Hh,
                              const float* as_, const float* ad_,
                              float* Out, int OutStride) {
    node_logits_kernel<<<nb((long)N * Hh, 8), TB, 0, stream>>>(Hlin, as_, ad_, SL, SD, N, Npad, Hh, C);
    fill_zero_kernel<<<nb((long)N * Hh, TB), TB, 0, stream>>>(Mx, (long)N * Hh);
    fill_zero_kernel<<<nb((long)N * Hh, TB), TB, 0, stream>>>((unsigned*)DEN, (long)N * Hh);
    edge_max_kernel<<<nb(Etot * Hh, TB), TB, 0, stream>>>(ei, E, Etot, SL, SD, Mx, Hh);
    edge_expsum_kernel<<<nb(Etot * Hh, TB), TB, 0, stream>>>(ei, E, Etot, SL, SD, Mx, DEN, EXb, Hh);
    edge_agg_kernel<<<nb(Etot, 8), TB, 0, stream>>>(ei, E, Etot, Hlin, Npad, nvalid, C, Hh,
                                                    EXb, DEN, Out, OutStride);
  };

  // ===== Layer 1: x[N,300] @ W1[300,256], H=2, C=128 =====
  cvt_node_bf16_kernel<<<nb((long)N * FINP, TB), TB, 0, stream>>>(x, Abuf, N, FIN, FINP);
  cvt_w_bf16t_kernel<<<nb((long)HID * FINP, TB), TB, 0, stream>>>(W1, Wb, FIN, HID, FINP, HID);
  { dim3 g(nb(N, 128), HID / 16);
    gemm_bf16_wmma_kernel<<<g, TB, 0, stream>>>(Abuf, Wb, Hbuf, N, HID, FINP); }
  fill_zero_kernel<<<nb((long)N * HID, TB), TB, 0, stream>>>((unsigned*)Gbuf, (long)N * HID);
  edge_softmax_agg(Hbuf, HID, HID, 128, 2, a1s, a1d, Gbuf, HID);
  bias_elu_cvt_kernel<<<nb((long)N * HID, TB), TB, 0, stream>>>(Gbuf, b1, Abuf, N, HID);

  // ===== Layer 2: h[N,256] @ W2[256,256], H=2, C=128 =====
  cvt_w_bf16t_kernel<<<nb((long)HID * HID, TB), TB, 0, stream>>>(W2, Wb, HID, HID, HID, HID);
  { dim3 g(nb(N, 128), HID / 16);
    gemm_bf16_wmma_kernel<<<g, TB, 0, stream>>>(Abuf, Wb, Hbuf, N, HID, HID); }
  fill_zero_kernel<<<nb((long)N * HID, TB), TB, 0, stream>>>((unsigned*)Gbuf, (long)N * HID);
  edge_softmax_agg(Hbuf, HID, HID, 128, 2, a2s, a2d, Gbuf, HID);
  bias_elu_cvt_kernel<<<nb((long)N * HID, TB), TB, 0, stream>>>(Gbuf, b2, Abuf, N, HID);

  // ===== Layer 3: h[N,256] @ W3[256,100], H=1, C=100 (mean over 1 head = identity) =====
  cvt_w_bf16t_kernel<<<nb((long)OUTP * HID, TB), TB, 0, stream>>>(W3, Wb, HID, OUT, HID, OUTP);
  { dim3 g(nb(N, 128), OUTP / 16);
    gemm_bf16_wmma_kernel<<<g, TB, 0, stream>>>(Abuf, Wb, Hbuf, N, OUTP, HID); }
  fill_zero_kernel<<<nb((long)out_size, TB), TB, 0, stream>>>((unsigned*)d_out, (long)out_size);
  edge_softmax_agg(Hbuf, OUTP, OUT, OUT, 1, a3s, a3d, (float*)d_out, OUT);
  bias_add_kernel<<<nb((long)N * OUT, TB), TB, 0, stream>>>((float*)d_out, b3, N, OUT);
}
